// DeepSeekBlock_65283502899671
// MI455X (gfx1250) — compile-verified, matrix operand order checked by reading, not run on hardware
//
#include <hip/hip_runtime.h>

#define DEVINL __device__ __forceinline__

constexpr int Hc    = 2048;
constexpr int NHc   = 16;
constexpr int NKVc  = 4;
constexpr int HDc   = 128;            // Hc / NHc
constexpr int FFc   = 8192;
constexpr int Bc    = 2;
constexpr int Sc    = 2048;
constexpr int Mrows = Bc * Sc;        // 4096 tokens
constexpr int KVD   = NKVc * HDc;     // 512
constexpr float RSCALE = 0.08838834764831845f;   // 1/sqrt(128)

typedef __attribute__((ext_vector_type(16))) __bf16       v16bf;
typedef __attribute__((ext_vector_type(8)))  float        v8f;
typedef __attribute__((ext_vector_type(4)))  unsigned int v4u;

struct FragU { v4u x; v4u y; };

DEVINL unsigned short f32_to_bf16(float f) {
  unsigned int u = __builtin_bit_cast(unsigned int, f);
  u = (u + 0x7FFFu + ((u >> 16) & 1u)) >> 16;     // round-to-nearest-even
  return (unsigned short)u;
}

// Load a 16x32 bf16 fragment (A-layout: rows in lanes, K packed along VGPRs;
// identical packing serves as B-fragment with "row" = N since W is [N,K]).
// lane l: row = row0 + (l&15); elems 0..7 -> K = k0 + (l>>4)*8 .. +7,
//         elems 8..15 -> K = k0 + 16 + (l>>4)*8 .. +7   (two b128 loads).
DEVINL v16bf load_frag(const unsigned short* p, int ld, int row0, int k0) {
  const int l = threadIdx.x & 31;
  const unsigned short* q = p + (size_t)(row0 + (l & 15)) * (size_t)ld
                              + (size_t)(k0 + ((l >> 4) << 3));
  FragU f;
  f.x = *(const v4u*)q;
  f.y = *(const v4u*)(q + 16);
  return __builtin_bit_cast(v16bf, f);
}

DEVINL v8f wmma_bf16(v16bf a, v16bf b, v8f c) {
  return __builtin_amdgcn_wmma_f32_16x16x32_bf16(
      /*neg_a=*/false, a, /*neg_b=*/false, b,
      /*c_mod=*/(short)0, c, /*reuse_a=*/false, /*reuse_b=*/false);
}

// ---------------------------------------------------------------- converts
__global__ void convert_bf16_kernel(const float* __restrict__ in,
                                    unsigned short* __restrict__ out, int n) {
  int i = (blockIdx.x * 256 + threadIdx.x) * 4;
  if (i + 3 < n) {
    float4 v = *(const float4*)(in + i);
    out[i + 0] = f32_to_bf16(v.x);
    out[i + 1] = f32_to_bf16(v.y);
    out[i + 2] = f32_to_bf16(v.z);
    out[i + 3] = f32_to_bf16(v.w);
  } else {
    for (; i < n; ++i) out[i] = f32_to_bf16(in[i]);
  }
}

// ---------------------------------------------------------------- RMSNorm
__global__ void __launch_bounds__(256)
rmsnorm_kernel(const float* __restrict__ x, const float* __restrict__ g,
               unsigned short* __restrict__ out) {
  __shared__ float red[8];
  const int row = blockIdx.x;
  const int t   = threadIdx.x;
  const float* xr = x + (size_t)row * Hc;
  float v[8];
  float s = 0.0f;
#pragma unroll
  for (int i = 0; i < 8; ++i) { v[i] = xr[t + i * 256]; s += v[i] * v[i]; }
#pragma unroll
  for (int m = 1; m < 32; m <<= 1) s += __shfl_xor(s, m, 32);
  if ((t & 31) == 0) red[t >> 5] = s;
  __syncthreads();
  if (t < 32) {
    float r = (t < 8) ? red[t] : 0.0f;
#pragma unroll
    for (int m = 1; m < 8; m <<= 1) r += __shfl_xor(r, m, 32);
    if (t == 0) red[0] = r;
  }
  __syncthreads();
  const float inv = rsqrtf(red[0] * (1.0f / Hc) + 1e-5f);
  unsigned short* orow = out + (size_t)row * Hc;
#pragma unroll
  for (int i = 0; i < 8; ++i) {
    const int c = t + i * 256;
    orow[c] = f32_to_bf16(v[i] * inv * g[c]);
  }
}

// ---------------------------------------------------------------- GEMM
// C[M,N] = A[M,K] (bf16, row-major) @ W[N,K]^T (bf16, torch-style) with
// fused epilogues. Block: 256 thr = 8 waves, 64x128 tile; wave: 32x32.
enum { GM_BF16 = 0, GM_ADDRES = 1, GM_BF16_VT = 2, GM_SILUMUL = 3 };

template <int MODE>
__global__ void __launch_bounds__(256)
gemm_wmma_kernel(const unsigned short* __restrict__ A,
                 const unsigned short* __restrict__ W,
                 const unsigned short* __restrict__ W2,
                 const float* __restrict__ Res,
                 void* __restrict__ OutP,
                 int Mx, int Nx, int Kx) {
  const int w  = threadIdx.x >> 5;
  const int m0 = blockIdx.y * 64  + (w >> 2) * 32;
  const int n0 = blockIdx.x * 128 + (w & 3) * 32;
  v8f zero = {};
  v8f acc00 = zero, acc01 = zero, acc10 = zero, acc11 = zero;
  v8f ac200 = zero, ac201 = zero, ac210 = zero, ac211 = zero;

  for (int k0 = 0; k0 < Kx; k0 += 32) {
    v16bf a0 = load_frag(A, Kx, m0,      k0);
    v16bf a1 = load_frag(A, Kx, m0 + 16, k0);
    v16bf b0 = load_frag(W, Kx, n0,      k0);
    v16bf b1 = load_frag(W, Kx, n0 + 16, k0);
    acc00 = wmma_bf16(a0, b0, acc00);
    acc01 = wmma_bf16(a0, b1, acc01);
    acc10 = wmma_bf16(a1, b0, acc10);
    acc11 = wmma_bf16(a1, b1, acc11);
    if constexpr (MODE == GM_SILUMUL) {
      v16bf c0 = load_frag(W2, Kx, n0,      k0);
      v16bf c1 = load_frag(W2, Kx, n0 + 16, k0);
      ac200 = wmma_bf16(a0, c0, ac200);
      ac201 = wmma_bf16(a0, c1, ac201);
      ac210 = wmma_bf16(a1, c0, ac210);
      ac211 = wmma_bf16(a1, c1, ac211);
    }
    if (k0 + 32 < Kx) {   // -> global_prefetch_b8 for the streamed tiles
      __builtin_prefetch(A + (size_t)m0 * Kx + k0 + 32, 0, 1);
      __builtin_prefetch(W + (size_t)n0 * Kx + k0 + 32, 0, 1);
    }
  }

  const int l  = threadIdx.x & 31;
  const int cb = l & 15;
  const int rb = (l >> 4) << 3;   // C-layout: lanes 16-31 hold rows M+8..M+15
  v8f* accs[4]  = { &acc00, &acc01, &acc10, &acc11 };
  v8f* accs2[4] = { &ac200, &ac201, &ac210, &ac211 };
#pragma unroll
  for (int t = 0; t < 4; ++t) {
    const int mr0 = m0 + (t >> 1) * 16 + rb;
    const int col = n0 + (t & 1) * 16 + cb;
#pragma unroll
    for (int r = 0; r < 8; ++r) {
      const size_t idx = (size_t)(mr0 + r) * (size_t)Nx + col;
      float v = (*accs[t])[r];
      if constexpr (MODE == GM_ADDRES) {
        ((float*)OutP)[idx] = v + Res[idx];
      } else if constexpr (MODE == GM_BF16) {
        ((unsigned short*)OutP)[idx] = f32_to_bf16(v);
      } else if constexpr (MODE == GM_BF16_VT) {
        // store V transposed per batch: Vt[(b*KVD + feature)*Sc + s]
        const int mr = mr0 + r;
        const int bb = mr >> 11;          // / Sc
        const int ss = mr & (Sc - 1);
        ((unsigned short*)OutP)[((size_t)(bb * KVD + col)) * Sc + ss] =
            f32_to_bf16(v);
      } else {  // GM_SILUMUL : silu(gate) * up -> bf16
        float gte = v, up = (*accs2[t])[r];
        float si  = gte / (1.0f + __expf(-gte));
        ((unsigned short*)OutP)[idx] = f32_to_bf16(si * up);
      }
    }
  }
}

// ---------------------------------------------------------------- Flash attention
// grid (Sc/128, B*NH); block 256 = 8 waves; wave owns 16 q rows, streams
// 32-key steps with online softmax. Causal mask computed analytically.
__global__ void __launch_bounds__(256)
flash_attn_kernel(const unsigned short* __restrict__ Q,     // [M, H] bf16
                  const unsigned short* __restrict__ Km,    // [M, KVD] bf16
                  const unsigned short* __restrict__ Vt,    // [B*KVD, Sc] bf16
                  unsigned short* __restrict__ O) {         // [M, H] bf16
  __shared__ __align__(16) unsigned short plds[8][16 * 32];
  const int bh  = blockIdx.y;
  const int b   = bh / NHc;
  const int h   = bh % NHc;
  const int kvh = h / (NHc / NKVc);   // repeat_interleave mapping
  const int w    = threadIdx.x >> 5;
  const int l    = threadIdx.x & 31;
  const int hsel = l >> 4, ln = l & 15;
  const int q0    = blockIdx.x * 128 + w * 16;
  const int qrow0 = b * Sc + q0;

  v16bf qf[4];
#pragma unroll
  for (int kk = 0; kk < 4; ++kk)
    qf[kk] = load_frag(Q, Hc, qrow0, h * HDc + kk * 32);

  v8f zero = {};
  v8f o[8];
#pragma unroll
  for (int nt = 0; nt < 8; ++nt) o[nt] = zero;
  float rm[8], rs[8];
#pragma unroll
  for (int r = 0; r < 8; ++r) { rm[r] = -__builtin_inff(); rs[r] = 0.0f; }

  const unsigned short* Vth = Vt + ((size_t)b * KVD + kvh * HDc) * (size_t)Sc;
  const int nsteps = (q0 + 16 + 31) >> 5;

  for (int j = 0; j < nsteps; ++j) {
    const int kbase = j * 32;
    // scores: 16 q rows x 32 keys, K-dim = HD = 128 -> 8 WMMAs
    v8f s0 = zero, s1 = zero;
#pragma unroll
    for (int kk = 0; kk < 4; ++kk) {
      v16bf k0f = load_frag(Km, KVD, b * Sc + kbase,      kvh * HDc + kk * 32);
      v16bf k1f = load_frag(Km, KVD, b * Sc + kbase + 16, kvh * HDc + kk * 32);
      s0 = wmma_bf16(qf[kk], k0f, s0);
      s1 = wmma_bf16(qf[kk], k1f, s1);
    }
    float p0[8], p1[8];
#pragma unroll
    for (int r = 0; r < 8; ++r) {
      const int qi = q0 + hsel * 8 + r;
      float x0 = s0[r] * RSCALE;
      float x1 = s1[r] * RSCALE;
      if (kbase + ln      > qi) x0 = -__builtin_inff();
      if (kbase + 16 + ln > qi) x1 = -__builtin_inff();
      float m2 = fmaxf(x0, x1);
#pragma unroll
      for (int msk = 1; msk < 16; msk <<= 1)
        m2 = fmaxf(m2, __shfl_xor(m2, msk, 32));   // 16-lane row reduce
      const float newm = fmaxf(rm[r], m2);
      const float corr = __expf(rm[r] - newm);
      const float e0 = __expf(x0 - newm);
      const float e1 = __expf(x1 - newm);
      p0[r] = e0; p1[r] = e1;
      float ps = e0 + e1;
#pragma unroll
      for (int msk = 1; msk < 16; msk <<= 1)
        ps += __shfl_xor(ps, msk, 32);
      rs[r] = rs[r] * corr + ps;
      rm[r] = newm;
#pragma unroll
      for (int nt = 0; nt < 8; ++nt) o[nt][r] *= corr;
    }
    // C-layout (q rows in VGPRs) -> A-layout (q rows in lanes) via LDS
#pragma unroll
    for (int r = 0; r < 8; ++r) {
      const int row = hsel * 8 + r;
      plds[w][row * 32 + ln]      = f32_to_bf16(p0[r]);
      plds[w][row * 32 + 16 + ln] = f32_to_bf16(p1[r]);
    }
#if __has_builtin(__builtin_amdgcn_s_wait_dscnt)
    __builtin_amdgcn_s_wait_dscnt(0);
#else
    asm volatile("s_wait_dscnt 0" ::: "memory");
#endif
    v16bf pf = load_frag(&plds[w][0], 32, 0, 0);
#pragma unroll
    for (int nt = 0; nt < 8; ++nt) {     // PV: N = 128 hd cols -> 8 WMMAs
      v16bf vb = load_frag(Vth, Sc, nt * 16, kbase);
      o[nt] = wmma_bf16(pf, vb, o[nt]);
    }
  }
#pragma unroll
  for (int nt = 0; nt < 8; ++nt) {
#pragma unroll
    for (int r = 0; r < 8; ++r) {
      const size_t orow = (size_t)qrow0 + hsel * 8 + r;
      O[orow * Hc + h * HDc + nt * 16 + ln] = f32_to_bf16(o[nt][r] / rs[r]);
    }
  }
}

// ---------------------------------------------------------------- launch
extern "C" void kernel_launch(void* const* d_in, const int* in_sizes, int n_in,
                              void* d_out, int out_size, void* d_ws, size_t ws_size,
                              hipStream_t stream) {
  (void)in_sizes; (void)n_in; (void)out_size; (void)ws_size;
  const float* x  = (const float*)d_in[0];
  // d_in[1] = attention_mask: standard causal mask, applied analytically.
  const float* wq = (const float*)d_in[2];
  const float* wk = (const float*)d_in[3];
  const float* wv = (const float*)d_in[4];
  const float* wo = (const float*)d_in[5];
  const float* n1 = (const float*)d_in[6];
  const float* n2 = (const float*)d_in[7];
  const float* wg = (const float*)d_in[8];
  const float* wu = (const float*)d_in[9];
  const float* wd = (const float*)d_in[10];
  float* out = (float*)d_out;

  char*  base = (char*)d_ws;
  size_t off  = 0;
  auto alloc = [&](size_t elems) -> unsigned short* {
    unsigned short* p = (unsigned short*)(base + off);
    off += ((elems * sizeof(unsigned short)) + 255) & ~(size_t)255;
    return p;
  };
  unsigned short* xn   = alloc((size_t)Mrows * Hc);
  unsigned short* xn2  = alloc((size_t)Mrows * Hc);
  unsigned short* qb   = alloc((size_t)Mrows * Hc);
  unsigned short* kb   = alloc((size_t)Mrows * KVD);
  unsigned short* vt   = alloc((size_t)Bc * KVD * Sc);
  unsigned short* attn = alloc((size_t)Mrows * Hc);
  unsigned short* gu   = alloc((size_t)Mrows * FFc);
  unsigned short* wqb  = alloc((size_t)Hc * Hc);
  unsigned short* wkb  = alloc((size_t)KVD * Hc);
  unsigned short* wvb  = alloc((size_t)KVD * Hc);
  unsigned short* wob  = alloc((size_t)Hc * Hc);
  unsigned short* wgb  = alloc((size_t)FFc * Hc);
  unsigned short* wub  = alloc((size_t)FFc * Hc);
  unsigned short* wdb  = alloc((size_t)Hc * FFc);

  auto cvt = [&](const float* src, unsigned short* dst, size_t n) {
    convert_bf16_kernel<<<dim3((unsigned)((n + 1023) / 1024)), 256, 0, stream>>>(
        src, dst, (int)n);
  };
  cvt(wq, wqb, (size_t)Hc * Hc);
  cvt(wk, wkb, (size_t)KVD * Hc);
  cvt(wv, wvb, (size_t)KVD * Hc);
  cvt(wo, wob, (size_t)Hc * Hc);
  cvt(wg, wgb, (size_t)FFc * Hc);
  cvt(wu, wub, (size_t)FFc * Hc);
  cvt(wd, wdb, (size_t)Hc * FFc);

  dim3 blk(256);
  // pre-attention norm
  rmsnorm_kernel<<<Mrows, blk, 0, stream>>>(x, n1, xn);
  // Q/K/V projections (V stored transposed per batch for the PV B-fragments)
  gemm_wmma_kernel<GM_BF16><<<dim3(Hc / 128, Mrows / 64), blk, 0, stream>>>(
      xn, wqb, nullptr, nullptr, qb, Mrows, Hc, Hc);
  gemm_wmma_kernel<GM_BF16><<<dim3(KVD / 128, Mrows / 64), blk, 0, stream>>>(
      xn, wkb, nullptr, nullptr, kb, Mrows, KVD, Hc);
  gemm_wmma_kernel<GM_BF16_VT><<<dim3(KVD / 128, Mrows / 64), blk, 0, stream>>>(
      xn, wvb, nullptr, nullptr, vt, Mrows, KVD, Hc);
  // causal GQA flash attention
  flash_attn_kernel<<<dim3(Sc / 128, Bc * NHc), blk, 0, stream>>>(qb, kb, vt, attn);
  // output projection + residual -> d_out (f32)
  gemm_wmma_kernel<GM_ADDRES><<<dim3(Hc / 128, Mrows / 64), blk, 0, stream>>>(
      attn, wob, nullptr, x, out, Mrows, Hc, Hc);
  // post-attention norm
  rmsnorm_kernel<<<Mrows, blk, 0, stream>>>(out, n2, xn2);
  // fused gate/up dual-GEMM with SiLU*up epilogue -> bf16
  gemm_wmma_kernel<GM_SILUMUL><<<dim3(FFc / 128, Mrows / 64), blk, 0, stream>>>(
      xn2, wgb, wub, nullptr, gu, Mrows, FFc, Hc);
  // down projection + residual accumulate into d_out
  gemm_wmma_kernel<GM_ADDRES><<<dim3(Hc / 128, Mrows / 64), blk, 0, stream>>>(
      gu, wdb, nullptr, out, out, Mrows, Hc, FFc);
}
